// RelationalGNNSelector_13932873908847
// MI455X (gfx1250) — matrix operand order repaired and checked by reading, not compiled, not run on hardware
//
#include <hip/hip_runtime.h>
#include <hip/hip_bf16.h>

// ---------------- problem constants ----------------
#define Bc   16
#define Lc   8
#define Hc   384
#define Wc   384
#define HWc  (Hc * Wc)        // 147456
#define SUMN (Bc * Lc)        // 128
#define HIDc 64
#define SLICES 8
#define PIX_PER_SLICE (HWc / SLICES)   // 18432
#define VEC_PER_SLICE (PIX_PER_SLICE / 4)  // 4608 float4 per slice

typedef _Float16 v16h __attribute__((ext_vector_type(16)));
typedef float    v8f  __attribute__((ext_vector_type(8)));

__device__ __forceinline__ float sigmoidf_(float x) {
    return 1.0f / (1.0f + __expf(-x));
}

// ---------------- block reduction (256 threads) ----------------
__device__ __forceinline__ float block_reduce256(float v, float* red) {
    int t = threadIdx.x;
    red[t] = v;
    __syncthreads();
    for (int s = 128; s > 0; s >>= 1) {
        if (t < s) red[t] += red[t + s];
        __syncthreads();
    }
    return red[0];
}

// ---------------- kernel 1: partial sums of conf = sigmoid(max(ch0,ch1)) ----------------
// Pure HBM-bandwidth kernel (151 MB total): use b128 loads.
__global__ void k_meanconf(const float* __restrict__ psm, float* __restrict__ mc_part) {
    __shared__ float red[256];
    int img = blockIdx.x;           // 0..127
    int sl  = blockIdx.y;           // 0..7
    const float4* c0 = (const float4*)(psm + (size_t)img * 2 * HWc + sl * PIX_PER_SLICE);
    const float4* c1 = (const float4*)(psm + (size_t)img * 2 * HWc + HWc + sl * PIX_PER_SLICE);
    float acc = 0.0f;
    for (int i = threadIdx.x; i < VEC_PER_SLICE; i += 256) {
        float4 a = c0[i];
        float4 b = c1[i];
        acc += sigmoidf_(fmaxf(a.x, b.x));
        acc += sigmoidf_(fmaxf(a.y, b.y));
        acc += sigmoidf_(fmaxf(a.z, b.z));
        acc += sigmoidf_(fmaxf(a.w, b.w));
    }
    float s = block_reduce256(acc, red);
    if (threadIdx.x == 0) mc_part[img * SLICES + sl] = s;
}

// ---------------- kernel 2: partial sums of demand = mean(req_mask[b, l=0]) ----------------
__global__ void k_demand(const float* __restrict__ req, float* __restrict__ dm_part) {
    __shared__ float red[256];
    int b  = blockIdx.x;            // 0..15
    int sl = blockIdx.y;            // 0..7
    const float4* Di = (const float4*)(req + (size_t)(b * Lc) * HWc + sl * PIX_PER_SLICE);
    float acc = 0.0f;
    for (int i = threadIdx.x; i < VEC_PER_SLICE; i += 256) {
        float4 d = Di[i];
        acc += (d.x + d.y) + (d.z + d.w);
    }
    float s = block_reduce256(acc, red);
    if (threadIdx.x == 0) dm_part[b * SLICES + sl] = s;
}

// conf tap with border-zero semantics; recomputed from psm (L2-resident: 151MB < 192MB L2)
__device__ __forceinline__ float conf_tap(const float* __restrict__ c0,
                                          const float* __restrict__ c1,
                                          int y, int x) {
    bool valid = (x >= 0) & (x < Wc) & (y >= 0) & (y < Hc);
    int yc = min(max(y, 0), Hc - 1);
    int xc = min(max(x, 0), Wc - 1);
    int idx = yc * Wc + xc;
    float v = sigmoidf_(fmaxf(c0[idx], c1[idx]));
    return valid ? v : 0.0f;
}

// ---------------- kernel 3: partial sums of overlap = mean(Di * warp(conf, Aij)) ----------------
__global__ void k_overlap(const float* __restrict__ psm, const float* __restrict__ req,
                          const float* __restrict__ na, float* __restrict__ ov_part) {
    __shared__ float red[256];
    int img = blockIdx.x;           // 0..127  (b*8 + l)
    int sl  = blockIdx.y;           // 0..7
    int b = img >> 3, l = img & 7;
    const float* A = na + (size_t)((b * Lc + 0) * Lc + l) * 6;   // norm_affine[b,0,l]
    float a00 = A[0], a01 = A[1], a02 = A[2];
    float a10 = A[3], a11 = A[4], a12 = A[5];
    const float* c0 = psm + (size_t)img * 2 * HWc;
    const float* c1 = c0 + HWc;
    const float* Di = req + (size_t)(b * Lc) * HWc;
    const float ginv  = 2.0f / (float)(Wc - 1);
    const float half1 = (float)(Wc - 1) * 0.5f;

    int base = sl * PIX_PER_SLICE;
    float acc = 0.0f;
    for (int p = base + threadIdx.x; p < base + PIX_PER_SLICE; p += 256) {
        int y = p / Wc;
        int x = p - y * Wc;
        float gx = -1.0f + (float)x * ginv;
        float gy = -1.0f + (float)y * ginv;
        float sx = a00 * gx + a01 * gy + a02;
        float sy = a10 * gx + a11 * gy + a12;
        float px = (sx + 1.0f) * half1;
        float py = (sy + 1.0f) * half1;
        float fx = floorf(px), fy = floorf(py);
        float wx = px - fx,    wy = py - fy;
        int x0 = (int)fx, y0 = (int)fy;
        float v00 = conf_tap(c0, c1, y0,     x0);
        float v01 = conf_tap(c0, c1, y0,     x0 + 1);
        float v10 = conf_tap(c0, c1, y0 + 1, x0);
        float v11 = conf_tap(c0, c1, y0 + 1, x0 + 1);
        float w = v00 * (1.0f - wy) * (1.0f - wx) + v01 * (1.0f - wy) * wx
                + v10 * wy * (1.0f - wx)          + v11 * wy * wx;
        acc += Di[p] * w;
    }
    float s = block_reduce256(acc, red);
    if (threadIdx.x == 0) ov_part[img * SLICES + sl] = s;
}

// ---------------- WMMA fragment loaders (CDNA5 16x16x32 f16 layouts, wave32) ----------------
// A (16x32 f16, MxK): lanes 0-15 -> M=lane, lanes 16-31 -> M=lane-16.
// Element s holds K = 16*h + s (matches B fragment K order).
// 16 contiguous f16 per lane.
__device__ __forceinline__ v16h load_a_frag(const _Float16* __restrict__ A, int lda,
                                            int row0, int k0, int lane) {
    int r = lane & 15, h = lane >> 4;
    const _Float16* base = A + (size_t)(row0 + r) * lda + k0;
    v16h a;
#pragma unroll
    for (int s = 0; s < 16; ++s) {
        a[s] = base[16 * h + s];
    }
    return a;
}

// B (32x16 f16, KxN): lanes 0-15 -> N=lane (K=0..15 across VGPRs),
// lanes 16-31 -> N=lane-16 (K=16..31).  VGPR j holds K = 16*h + 2j, +1.
__device__ __forceinline__ v16h load_b_frag(const _Float16* __restrict__ W, int ldw,
                                            int k0, int col0, int lane) {
    int n = lane & 15, h = lane >> 4;
    const _Float16* base = W + (size_t)(k0 + 16 * h) * ldw + col0 + n;
    v16h b;
#pragma unroll
    for (int j = 0; j < 8; ++j) {
        b[2 * j]     = base[(2 * j) * ldw];
        b[2 * j + 1] = base[(2 * j + 1) * ldw];
    }
    return b;
}

// D = act( A[MxK] @ W[KxN] + bias ), f16 in, f32 accumulate via v_wmma_f32_16x16x32_f16,
// f16 out to LDS.  Single wave executes all tiles (EXEC all ones at WMMA).
__device__ __forceinline__ void gemm_wmma(const _Float16* __restrict__ A, int lda,
                                          const _Float16* __restrict__ W, int ldw,
                                          const float* __restrict__ bias, bool relu,
                                          _Float16* __restrict__ D, int ldd,
                                          int M, int N, int K, int lane) {
    int n = lane & 15, h = lane >> 4;
    for (int mt = 0; mt < M; mt += 16) {
        for (int nt = 0; nt < N; nt += 16) {
            v8f c = {0.f, 0.f, 0.f, 0.f, 0.f, 0.f, 0.f, 0.f};
            for (int kt = 0; kt < K; kt += 32) {
                v16h a = load_a_frag(A, lda, mt, kt, lane);
                v16h b = load_b_frag(W, ldw, kt, nt, lane);
                c = __builtin_amdgcn_wmma_f32_16x16x32_f16(
                        /*neg_a=*/false, a, /*neg_b=*/false, b,
                        /*c_mod=*/(short)0, c, /*reuse_a=*/false, /*reuse_b=*/false);
            }
            int col = nt + n;
            float bv = bias[col];
#pragma unroll
            for (int r = 0; r < 8; ++r) {
                float v = c[r] + bv;
                if (relu) v = fmaxf(v, 0.0f);
                D[(size_t)(mt + r + 8 * h) * ldd + col] = (_Float16)v;
            }
        }
    }
}

// ---------------- kernel 4: feature build + all MLPs (WMMA) + output ----------------
__global__ void __launch_bounds__(32) k_mlp(
    const float* __restrict__ na,
    const float* __restrict__ mc_part, const float* __restrict__ dm_part,
    const float* __restrict__ ov_part,
    const float* __restrict__ qW1, const float* __restrict__ qb1,
    const float* __restrict__ qW2, const float* __restrict__ qb2,
    const float* __restrict__ kW1, const float* __restrict__ kb1,
    const float* __restrict__ kW2, const float* __restrict__ kb2,
    const float* __restrict__ eW1, const float* __restrict__ eb1,
    const float* __restrict__ eW2, const float* __restrict__ eb2,
    float* __restrict__ out)
{
    __shared__ _Float16 sQK[128 * 128];   // 32 KB: qk matrix (also stages feat_j in cols 0..31)
    __shared__ _Float16 sW [128 * 64];    // 16 KB: current weight matrix
    __shared__ _Float16 sH [128 * 64];    // 16 KB: hidden activations (tail rows reused for q)
    const int lane = threadIdx.x;
    const float invHW = 1.0f / (float)HWc;

    // ---- stage feat_j (128 x 8, zero-padded to K=32) into sQK cols 0..31 ----
    for (int row = lane; row < 128; row += 32) {
        int b = row >> 3, l = row & 7;
        float mc = 0.f, ov = 0.f, dm = 0.f;
#pragma unroll
        for (int s = 0; s < SLICES; ++s) {
            mc += mc_part[row * SLICES + s];
            ov += ov_part[row * SLICES + s];
            dm += dm_part[b * SLICES + s];
        }
        mc *= invHW; ov *= invHW; dm *= invHW;
        const float* Aij = na + (size_t)((b * Lc + 0) * Lc + l) * 6;
        float dx = Aij[2], dy = Aij[5];
        float dist = sqrtf(dx * dx + dy * dy);
        // cosd/sind of (yaw0 - yawl) without atan2: yaw from diag entries (cos,sin) ~ (d00,d10)/r
        const float* dgl = na + (size_t)((b * Lc + l) * Lc + l) * 6;
        const float* dg0 = na + (size_t)((b * Lc + 0) * Lc + 0) * 6;
        float cl = dgl[0], slv = dgl[3];
        float c0 = dg0[0], s0v = dg0[3];
        float rl = rsqrtf(cl * cl + slv * slv + 1e-30f);
        float r0 = rsqrtf(c0 * c0 + s0v * s0v + 1e-30f);
        float cosd = (c0 * cl + s0v * slv) * r0 * rl;
        float sind = (s0v * cl - c0 * slv) * r0 * rl;
        _Float16* dst = &sQK[(size_t)row * 128];
        dst[0] = (_Float16)mc;   dst[1] = (_Float16)ov;
        dst[2] = (_Float16)dx;   dst[3] = (_Float16)dy;
        dst[4] = (_Float16)cosd; dst[5] = (_Float16)sind;
        dst[6] = (_Float16)dist; dst[7] = (_Float16)dm;
        for (int j = 8; j < 32; ++j) dst[j] = (_Float16)0.0f;
    }
    // ---- kW1 (8x64) zero-padded to 32x64 ----
    for (int idx = lane; idx < 32 * 64; idx += 32) {
        int r = idx >> 6, c = idx & 63;
        sW[idx] = (r < 8) ? (_Float16)kW1[r * 64 + c] : (_Float16)0.0f;
    }
    __syncthreads();
    // hidden_k = relu(feat_j @ kW1 + kb1)  -> sH (128x64)
    gemm_wmma(sQK, 128, sW, 64, kb1, true, sH, 64, 128, 64, 32, lane);
    __syncthreads();
    for (int idx = lane; idx < 64 * 64; idx += 32) sW[idx] = (_Float16)kW2[idx];
    __syncthreads();
    // k = hidden_k @ kW2 + kb2  -> sQK cols 64..127
    gemm_wmma(sH, 64, sW, 64, kb2, false, sQK + 64, 128, 128, 64, 64, lane);
    __syncthreads();

    // ---- q-MLP (16 rows).  Use sH rows 112..127 as 16x64 scratch ("sQ"). ----
    _Float16* sQ = sH + 112 * 64;
    for (int idx = lane; idx < 16 * 64; idx += 32) sQ[idx] = (_Float16)0.0f;
    __syncthreads();
    for (int b = lane; b < 16; b += 32) {
        float mc0 = 0.f, dm = 0.f;
#pragma unroll
        for (int s = 0; s < SLICES; ++s) {
            mc0 += mc_part[(b * Lc) * SLICES + s];
            dm  += dm_part[b * SLICES + s];
        }
        mc0 *= invHW; dm *= invHW;
        _Float16* dst = &sQ[(size_t)b * 64];
        dst[0] = (_Float16)mc0; dst[1] = (_Float16)dm;
        dst[4] = (_Float16)1.0f; dst[7] = (_Float16)dm;
    }
    for (int idx = lane; idx < 32 * 64; idx += 32) {
        int r = idx >> 6, c = idx & 63;
        sW[idx] = (r < 8) ? (_Float16)qW1[r * 64 + c] : (_Float16)0.0f;
    }
    __syncthreads();
    // hidden_q = relu(ego @ qW1 + qb1) -> sH rows 0..15
    gemm_wmma(sQ, 64, sW, 64, qb1, true, sH, 64, 16, 64, 32, lane);
    __syncthreads();
    for (int idx = lane; idx < 64 * 64; idx += 32) sW[idx] = (_Float16)qW2[idx];
    __syncthreads();
    // q = hidden_q @ qW2 + qb2 -> sQ (16x64)
    gemm_wmma(sH, 64, sW, 64, qb2, false, sQ, 64, 16, 64, 64, lane);
    __syncthreads();
    // broadcast q over l into sQK cols 0..63
    for (int idx = lane; idx < 128 * 64; idx += 32) {
        int row = idx >> 6, c = idx & 63;
        sQK[(size_t)row * 128 + c] = sQ[(size_t)(row >> 3) * 64 + c];
    }
    // eW1 (128x64)
    for (int idx = lane; idx < 128 * 64; idx += 32) sW[idx] = (_Float16)eW1[idx];
    __syncthreads();
    // hidden_e = relu(qk @ eW1 + eb1) -> sH (128x64)
    gemm_wmma(sQK, 128, sW, 64, eb1, true, sH, 64, 128, 64, 128, lane);
    __syncthreads();

    // ---- final: logits = hidden_e @ eW2 + eb2 ; logits[:,0] = NEG ; sigmoid+clip ----
    for (int row = lane; row < 128; row += 32) {
        float acc = eb2[0];
#pragma unroll 8
        for (int j = 0; j < 64; ++j) acc += (float)sH[(size_t)row * 64 + j] * eW2[j];
        float logit = ((row & 7) == 0) ? -1.0e9f : acc;
        float sg = 1.0f / (1.0f + __expf(-logit));
        out[row] = fminf(fmaxf(sg, 0.0f), 1.0f);
    }
}

// ---------------- host launcher ----------------
extern "C" void kernel_launch(void* const* d_in, const int* in_sizes, int n_in,
                              void* d_out, int out_size, void* d_ws, size_t ws_size,
                              hipStream_t stream) {
    const float* psm = (const float*)d_in[0];
    const float* req = (const float*)d_in[1];
    const float* na  = (const float*)d_in[2];
    // d_in[3] = record_len (unused: always full L)
    const float* qW1 = (const float*)d_in[4];
    const float* qb1 = (const float*)d_in[5];
    const float* qW2 = (const float*)d_in[6];
    const float* qb2 = (const float*)d_in[7];
    const float* kW1 = (const float*)d_in[8];
    const float* kb1 = (const float*)d_in[9];
    const float* kW2 = (const float*)d_in[10];
    const float* kb2 = (const float*)d_in[11];
    const float* eW1 = (const float*)d_in[12];
    const float* eb1 = (const float*)d_in[13];
    const float* eW2 = (const float*)d_in[14];
    const float* eb2 = (const float*)d_in[15];
    float* out = (float*)d_out;

    // workspace layout (floats): mc_part[128*8] | ov_part[128*8] | dm_part[16*8]
    float* ws      = (float*)d_ws;
    float* mc_part = ws;                 // 1024
    float* ov_part = ws + 1024;          // 1024
    float* dm_part = ws + 2048;          // 128
    // every slot is fully overwritten each call -> deterministic, no init needed

    k_meanconf<<<dim3(128, SLICES), 256, 0, stream>>>(psm, mc_part);
    k_demand  <<<dim3(16,  SLICES), 256, 0, stream>>>(req, dm_part);
    k_overlap <<<dim3(128, SLICES), 256, 0, stream>>>(psm, req, na, ov_part);
    k_mlp<<<1, 32, 0, stream>>>(na, mc_part, dm_part, ov_part,
                                qW1, qb1, qW2, qb2, kW1, kb1, kW2, kb2,
                                eW1, eb1, eW2, eb2, out);
}
